// PBatchLinear_84748294685035
// MI455X (gfx1250) — matrix operand order repaired and checked by reading, not compiled, not run on hardware
//
#include <hip/hip_runtime.h>
#include <hip/hip_fp16.h>

typedef __attribute__((ext_vector_type(16))) _Float16 v16h;
typedef __attribute__((ext_vector_type(8)))  float    v8f;

#define IN_F   8192
#define OUT_F  8192
#define KSPLIT 8
#define KCHUNK (IN_F / KSPLIT)          /* 1024 */
#define WAVES  8                        /* waves per block (wave32) */
#define ROWS_PER_BLOCK (WAVES * 16)     /* 128 */
#define ROW_BLOCKS (OUT_F / ROWS_PER_BLOCK)  /* 64 */

// ---------------------------------------------------------------------------
// Quantization parameters from global max|W|:
//   m = floor(maxabs * 2^16); max_place = max(ceil(log2 m)+1, 4)
//   step = 2^(max_place-4);  Wq = sign(w)*clip(round(v/step),1,15)*step/2^16
// ---------------------------------------------------------------------------
__device__ __forceinline__ void qparams(float maxabs, float& invStep, float& stepOverScale) {
    float m  = floorf(maxabs * 65536.0f);
    float mp = 4.0f;
    if (m > 0.0f) mp = fmaxf(ceilf(log2f(m)) + 1.0f, 4.0f);
    invStep       = exp2f(4.0f - mp);        // 1/step (exact power of 2)
    stepOverScale = exp2f(mp - 20.0f);       // step / 2^16
}

__device__ __forceinline__ float quant1(float w, float invStep) {
    float v = floorf(fabsf(w) * 65536.0f);           // |trunc(w*scale)| (exact)
    float q = floorf(fmaf(v, invStep, 0.5f));        // round-half-up(v/step)
    q = fminf(fmaxf(q, 1.0f), 15.0f);                // clip to [1,15]
    q = copysignf(q, w);
    return (w == 0.0f) ? 0.0f : q;                   // sign(0) == 0 in reference
}

// ---------------------------------------------------------------------------
// Kernel 0: zero the atomic-max slot (d_ws is not re-poisoned between replays)
// ---------------------------------------------------------------------------
__global__ void pb_init(unsigned* wsmax) { *wsmax = 0u; }

// ---------------------------------------------------------------------------
// Kernel 1: global max|W| (deterministic: uint atomicMax on non-negative floats)
// ---------------------------------------------------------------------------
__global__ __launch_bounds__(256) void pb_maxabs(const float* __restrict__ W,
                                                 unsigned* __restrict__ wsmax, int n4) {
    const float4* W4 = (const float4*)W;
    float m = 0.0f;
    for (int i = blockIdx.x * blockDim.x + threadIdx.x; i < n4;
         i += gridDim.x * blockDim.x) {
        float4 w = W4[i];
        m = fmaxf(m, fmaxf(fmaxf(fabsf(w.x), fabsf(w.y)),
                           fmaxf(fabsf(w.z), fabsf(w.w))));
    }
    for (int off = 16; off > 0; off >>= 1)
        m = fmaxf(m, __shfl_xor(m, off, 32));
    if ((threadIdx.x & 31) == 0) atomicMax(wsmax, __float_as_uint(m));
}

// ---------------------------------------------------------------------------
// Kernel 2: quantize-on-the-fly WMMA GEMV partials.
//   grid = (ROW_BLOCKS, KSPLIT); block = 256 (8 waves).
//   Wave w handles rows [rowBlock*128 + w*16, +16), K range [ks*1024, +1024).
//   A tile: 16x32 quantized weights (f16, exact integers |q|<=15).
//   B tile: 32x16 f16; col0 = x_hi, col1 = x_lo (hi/lo split ~ fp32 accuracy),
//   cols 2..15 read a zero-filled LDS row. out[m] = D[m,0]+D[m,1] via one
//   ds_swizzle (SWAPX1) exchange.
// ---------------------------------------------------------------------------
__global__ __launch_bounds__(256) void pb_wmma_gemv(const float* __restrict__ W,
                                                    const float* __restrict__ x,
                                                    const unsigned* __restrict__ wsmax,
                                                    float* __restrict__ partial) {
    // [0]=hi, [1]=lo, [2]=zeros : 6 KB, rows 32B-aligned for v16h loads
    __shared__ alignas(32) _Float16 xstage[3][KCHUNK];

    const int rowBlock = blockIdx.x;
    const int ks       = blockIdx.y;
    const int kbase    = ks * KCHUNK;

    // Stage hi/lo split of this K-chunk of x into LDS (uniform trip count).
    for (int i = threadIdx.x; i < KCHUNK; i += blockDim.x) {
        float xv = x[kbase + i];
        _Float16 h = (_Float16)xv;
        xstage[0][i] = h;
        xstage[1][i] = (_Float16)(xv - (float)h);
        xstage[2][i] = (_Float16)0.0f;
    }
    __syncthreads();

    float invStep, stepOverScale;
    qparams(__uint_as_float(*wsmax), invStep, stepOverScale);

    const int wave = threadIdx.x >> 5;
    const int lane = threadIdx.x & 31;
    const int col  = lane & 15;          // N column this lane owns in B/C/D
    const int half = lane >> 4;          // lane half selects K sub-ranges
    const int row  = rowBlock * ROWS_PER_BLOCK + wave * 16 + col;   // A: M = lane%16
    const float* wrow = W + (size_t)row * IN_F;
    const int koff = half * 16;          // A: half0 -> K 0-15, half1 -> K 16-31

    const _Float16* xsel = xstage[(col < 2) ? col : 2];

    v8f acc = {};
    for (int kk = 0; kk < KCHUNK; kk += 32) {
        const int k0 = kbase + kk;
        // ---- A: 16 fp32 weights per lane, two 32B contiguous chunks ----
        const float4* p0 = (const float4*)(wrow + k0 + koff);
        const float4* p1 = (const float4*)(wrow + k0 + 8 + koff);
        float4 a0 = p0[0], a1 = p0[1];
        float4 a2 = p1[0], a3 = p1[1];
        // Speculative prefetch of the next A tile (silently dropped if OOB).
        __builtin_prefetch(wrow + k0 + 32 + koff, 0, 1);
        __builtin_prefetch(wrow + k0 + 48 + koff, 0, 1);

        v16h A;
        A[0]  = (_Float16)quant1(a0.x, invStep);
        A[1]  = (_Float16)quant1(a0.y, invStep);
        A[2]  = (_Float16)quant1(a0.z, invStep);
        A[3]  = (_Float16)quant1(a0.w, invStep);
        A[4]  = (_Float16)quant1(a1.x, invStep);
        A[5]  = (_Float16)quant1(a1.y, invStep);
        A[6]  = (_Float16)quant1(a1.z, invStep);
        A[7]  = (_Float16)quant1(a1.w, invStep);
        A[8]  = (_Float16)quant1(a2.x, invStep);
        A[9]  = (_Float16)quant1(a2.y, invStep);
        A[10] = (_Float16)quant1(a2.z, invStep);
        A[11] = (_Float16)quant1(a2.w, invStep);
        A[12] = (_Float16)quant1(a3.x, invStep);
        A[13] = (_Float16)quant1(a3.y, invStep);
        A[14] = (_Float16)quant1(a3.z, invStep);
        A[15] = (_Float16)quant1(a3.w, invStep);

        // ---- B: 16 contiguous f16 from LDS (lanes<16: K 0-15, else K 16-31),
        //      columns >=2 read the zero row -> no VALU fixup needed ----
        v16h Bh = *(const v16h*)(xsel + kk + half * 16);   // 32B aligned

        acc = __builtin_amdgcn_wmma_f32_16x16x32_f16(
            /*neg_a=*/false, A, /*neg_b=*/false, Bh,
            /*c_mod=*/(short)0, acc, /*reuse_a=*/false, /*reuse_b=*/false);
    }

    // out[m] = D[m,0] + D[m,1]; one SWAPX1 lane exchange (xor 1) per register.
    // ds_swizzle group-of-32: offset[14:10]=xor=1, [4:0]=and=0x1F  -> 0x041F
    #pragma unroll
    for (int i = 0; i < 8; ++i) {
        int t = __builtin_amdgcn_ds_swizzle(__float_as_int(acc[i]), 0x041F);
        acc[i] += __int_as_float(t);
    }

    // C/D layout: half0 VGPR i -> M=i, half1 VGPR i -> M=8+i.
    if (col == 0) {
        const int rbase = rowBlock * ROWS_PER_BLOCK + wave * 16 + half * 8;
        #pragma unroll
        for (int i = 0; i < 8; ++i)
            partial[(size_t)ks * OUT_F + rbase + i] = acc[i] * stepOverScale;
    }
}

// ---------------------------------------------------------------------------
// Kernel 3: out[r] = bias[r] + sum_ks partial[ks][r]   (fixed order -> deterministic)
// ---------------------------------------------------------------------------
__global__ __launch_bounds__(256) void pb_finalize(const float* __restrict__ partial,
                                                   const float* __restrict__ bias,
                                                   float* __restrict__ out) {
    int r = blockIdx.x * blockDim.x + threadIdx.x;
    if (r < OUT_F) {
        float s = bias[r];
        #pragma unroll
        for (int ks = 0; ks < KSPLIT; ++ks)
            s += partial[(size_t)ks * OUT_F + r];
        out[r] = s;
    }
}

extern "C" void kernel_launch(void* const* d_in, const int* in_sizes, int n_in,
                              void* d_out, int out_size, void* d_ws, size_t ws_size,
                              hipStream_t stream) {
    (void)in_sizes; (void)n_in; (void)out_size; (void)ws_size;
    const float* x    = (const float*)d_in[0];   // input  [8192]
    const float* W    = (const float*)d_in[1];   // weight [8192,8192]
    const float* bias = (const float*)d_in[2];   // bias   [8192]
    float* out = (float*)d_out;

    unsigned* wsmax = (unsigned*)d_ws;                       // 4B max slot
    float* partial  = (float*)((char*)d_ws + 256);           // KSPLIT*8192 fp32 = 256KB

    pb_init<<<1, 1, 0, stream>>>(wsmax);
    pb_maxabs<<<2048, 256, 0, stream>>>(W, wsmax, (IN_F * OUT_F) / 4);
    dim3 grid(ROW_BLOCKS, KSPLIT);
    pb_wmma_gemv<<<grid, 256, 0, stream>>>(W, x, wsmax, partial);
    pb_finalize<<<(OUT_F + 255) / 256, 256, 0, stream>>>(partial, bias, out);
}
